// DenoisingAutoEncoder_60662118088764
// MI455X (gfx1250) — matrix operand order repaired and checked by reading, not compiled, not run on hardware
//
#include <hip/hip_runtime.h>
#include <stddef.h>

// ---------------------------------------------------------------------------
// Seq2seq LSTM autoencoder on MI455X (gfx1250).
//   V=32000 E=256 H=512 Ld=512 NL=2 S=1024 T=512
// - WMMA GEMMs (v_wmma_f32_16x16x32_f16) for the time-parallel input
//   projections of both encoder layers.
// - Persistent 16-WG cluster kernels (s_cluster_barrier) for the sequential
//   LSTM recurrences and the fused decoder (recurrence + 32000x512 logits +
//   packed-u64 atomic argmax + token feedback).
// - Tensor Data Mover (tensor_load_to_lds + s_wait_tensorcnt) DMAs the
//   LDS-resident Whh slices (4 contiguous 32KB chunks per matrix per WG).
// - All big weights converted once to f16; whole model fits in 192MB L2, so
//   HBM is touched ~once (~150MB ~= 7us at 23.3TB/s); the workload itself is
//   serialization-latency bound, which the cluster design minimizes.
// ---------------------------------------------------------------------------

typedef __attribute__((ext_vector_type(16))) _Float16 v16h;
typedef __attribute__((ext_vector_type(8)))  _Float16 v8h;
typedef __attribute__((ext_vector_type(8)))  float    v8f;

typedef __attribute__((ext_vector_type(4))) unsigned int tdm_u32x4;
typedef __attribute__((ext_vector_type(8))) int          tdm_i32x8;
typedef __attribute__((ext_vector_type(4))) int          tdm_i32x4;

#define Sdim 1024
#define Tdim 512
#define Vdim 32000
#define Edim 256
#define Hdim 512
#define G4H  2048   // 4*H = 4*Ld

__device__ __forceinline__ float sigm(float x) { return 1.0f / (1.0f + __expf(-x)); }

__device__ __forceinline__ float gload(const float* p) {
    return __hip_atomic_load(p, __ATOMIC_RELAXED, __HIP_MEMORY_SCOPE_AGENT);
}
__device__ __forceinline__ void gstore(float* p, float v) {
    __hip_atomic_store(p, v, __ATOMIC_RELAXED, __HIP_MEMORY_SCOPE_AGENT);
}

// ---------------------------------------------------------------------------
// TDM: DMA `n16` contiguous 16-bit elements from global -> LDS (1-row tile).
// D# per CDNA5 ISA. Issued once per wave (EXEC ignored),
// tracked by TENSORcnt. lds_off is a byte offset into the workgroup's LDS
// (dynamic segment starts at 0 in these kernels: no static __shared__).
__device__ __forceinline__ void tdm_load_1d(unsigned lds_off, const void* gptr, unsigned n16) {
    unsigned long long ga = (unsigned long long)(size_t)gptr;
    tdm_u32x4 g0;
    g0[0] = 1u;                                             // count=1 (valid), user mode
    g0[1] = lds_off;                                        // lds_addr [63:32]
    g0[2] = (unsigned)(ga & 0xFFFFFFFFull);                 // global_addr [95:64]
    g0[3] = (unsigned)((ga >> 32) & 0x01FFFFFFull)          // global_addr [120:96]
            | (2u << 30);                                   // type=2 ("image")
    tdm_i32x8 g1;
    g1[0] = 0x00010000;                                     // wg_mask=0, data_size=1 (2B)
    g1[1] = (int)((n16 & 0xFFFFu) << 16);                   // tensor_dim0[15:0] @ bits63:48
    g1[2] = (int)(((n16 >> 16) & 0xFFFFu) | (1u << 16));    // tensor_dim0[31:16], tensor_dim1=1
    g1[3] = (int)((n16 & 0xFFFFu) << 16);                   // tile_dim0 @ bits127:112
    g1[4] = 1;                                              // tile_dim1=1, tile_dim2=0
    g1[5] = (int)n16;                                       // tensor_dim0_stride[31:0]
    g1[6] = 0;
    g1[7] = 0;
    tdm_i32x4 gz = {0, 0, 0, 0};                            // groups 2/3 unused (<=2D)
#if __clang_major__ >= 23
    tdm_i32x8 gz8 = {0, 0, 0, 0, 0, 0, 0, 0};
    __builtin_amdgcn_tensor_load_to_lds(g0, g1, gz, gz, gz8, 0);
#else
    __builtin_amdgcn_tensor_load_to_lds(g0, g1, gz, gz, 0);
#endif
}

// ---------------------------------------------------------------------------
// Generic f32 -> f16 conversion
__global__ __launch_bounds__(256) void k_f32_to_f16(const float* __restrict__ in,
                                                    _Float16* __restrict__ out, int n) {
    int i = blockIdx.x * 256 + threadIdx.x;
    if (i < n) out[i] = (_Float16)in[i];
}

// Embedding gather, output f16 (GEMM A-matrix)
__global__ __launch_bounds__(256) void k_embed(const int* __restrict__ x,
                                               const float* __restrict__ emb,
                                               _Float16* __restrict__ em16) {
    int i = blockIdx.x * 256 + threadIdx.x;
    if (i < Sdim * Edim) {
        int s = i / Edim, e = i - s * Edim;
        em16[i] = (_Float16)emb[(size_t)x[s] * Edim + e];
    }
}

__global__ __launch_bounds__(256) void k_zero_f32(float* p, int n) {
    int i = blockIdx.x * 256 + threadIdx.x;
    if (i < n) p[i] = 0.0f;
}

// ---------------------------------------------------------------------------
// WMMA GEMM:  P[M][N] = A16[M][K] * W16[N][K]^T   (f16 in, f32 accumulate)
// One 16x16 tile per wave, 8 waves per workgroup. K multiple of 32.
// Fragment layouts per CDNA5 ISA 7.12.2 (wave32).
__global__ __launch_bounds__(256) void k_gemm_wmma(const _Float16* __restrict__ A,
                                                   const _Float16* __restrict__ W,
                                                   float* __restrict__ P,
                                                   int M, int N, int K) {
    int lane = threadIdx.x & 31;
    int wave = threadIdx.x >> 5;
    int tiles_n = N >> 4;
    int tile = blockIdx.x * 8 + wave;
    int tm = (tile / tiles_n) << 4;
    int tn = (tile - (tile / tiles_n) * tiles_n) << 4;
    if (tm >= M) return;                 // wave-uniform

    int hs  = lane >> 4;                 // half-wave select
    int l16 = lane & 15;

    v8f c = {};
    const _Float16* arow = A + (size_t)(tm + l16) * K;   // A row = M index
    const _Float16* wrow = W + (size_t)(tn + l16) * K;   // W row = B column

    for (int kb = 0; kb < K; kb += 32) {
        union { v16h v; v8h h[2]; } a;
        a.h[0] = *(const v8h*)(arow + kb + hs * 8);          // K = kb+{0..7|8..15}
        a.h[1] = *(const v8h*)(arow + kb + 16 + hs * 8);     // K = kb+16+{...}
        union { v16h v; v8h h[2]; } bu;
        bu.h[0] = *(const v8h*)(wrow + kb + hs * 8);         // K = kb+{0..7|8..15}
        bu.h[1] = *(const v8h*)(wrow + kb + 16 + hs * 8);    // K = kb+16+{...}
        v16h b = bu.v;
        c = __builtin_amdgcn_wmma_f32_16x16x32_f16(false, a.v, false, b,
                                                   (short)0, c, false, false);
    }
    // D layout: VGPR r -> row tm + r + hs*8, col tn + l16
#pragma unroll
    for (int r = 0; r < 8; ++r)
        P[(size_t)(tm + r + hs * 8) * N + tn + l16] = c[r];
}

// ---------------------------------------------------------------------------
// Persistent cluster LSTM recurrence (one encoder layer).
//   16 WGs x 256 threads (intended as one workgroup cluster).
//   WG w owns hidden units [w*32, w*32+32) -> 128 gate rows; Whh slice is
//   TDM-DMA'd into LDS as 4 contiguous 32KB chunks (one per gate block).
//   h shared cluster-wide via double-buffered global (L2) + s_cluster_barrier.
__global__ __launch_bounds__(256) void k_lstm_rec(const float* __restrict__ P,      // [S][4H] pre-gates
                                                  const _Float16* __restrict__ Whh, // [4H][H] f16
                                                  const float* __restrict__ b,      // [4H]
                                                  float* __restrict__ hbuf,         // [2][H]
                                                  _Float16* __restrict__ Hout16,    // [S][H]
                                                  float* __restrict__ hfinal,       // [H] or null
                                                  int S_) {
    constexpr int H = Hdim, UP = 32;
    extern __shared__ char smem[];
    _Float16* wlds = (_Float16*)smem;                         // 128*512 halfs @ LDS offset 0
    float* hcur = (float*)(smem + 128 * H * 2);               // 512
    float* part = hcur + H;                                   // 256
    float* cst  = part + 256;                                 // 32

    int w = blockIdx.x, tid = threadIdx.x;

    // TDM preload: gate block g -> 32 contiguous rows (g*H + w*32 ...) = 32KB
    if (tid < 32) {
#pragma unroll
        for (int g = 0; g < 4; ++g)
            tdm_load_1d((unsigned)(g * UP * H * 2),
                        Whh + ((size_t)g * H + (size_t)w * UP) * H,
                        (unsigned)(UP * H));
        __builtin_amdgcn_s_wait_tensorcnt(0);
    }
    if (tid < UP) { cst[tid] = 0.0f; gstore(&hbuf[w * UP + tid], 0.0f); }
    __threadfence();
    __syncthreads();
    __builtin_amdgcn_s_cluster_barrier();

    int lr = tid >> 1, kh = tid & 1;

    for (int t = 0; t < S_; ++t) {
        const float* hprev = hbuf + (size_t)(t & 1) * H;
        for (int i = tid; i < H; i += 256) hcur[i] = gload(&hprev[i]);
        __syncthreads();

        // 2 threads per gate row, 256 MACs each
        float acc = 0.0f;
        const _Float16* wr = wlds + lr * H + kh * 256;
        const float* hh = hcur + kh * 256;
#pragma unroll 8
        for (int k = 0; k < 256; ++k) acc += (float)wr[k] * hh[k];
        part[tid] = acc;
        __syncthreads();

        if (tid < UP) {
            int u = w * UP + tid;
            size_t pb = (size_t)t * G4H;
            float gi = P[pb + 0 * H + u] + b[0 * H + u] + part[(0 * UP + tid) * 2] + part[(0 * UP + tid) * 2 + 1];
            float gf = P[pb + 1 * H + u] + b[1 * H + u] + part[(1 * UP + tid) * 2] + part[(1 * UP + tid) * 2 + 1];
            float gg = P[pb + 2 * H + u] + b[2 * H + u] + part[(2 * UP + tid) * 2] + part[(2 * UP + tid) * 2 + 1];
            float go = P[pb + 3 * H + u] + b[3 * H + u] + part[(3 * UP + tid) * 2] + part[(3 * UP + tid) * 2 + 1];
            float c2 = sigm(gf) * cst[tid] + sigm(gi) * tanhf(gg);
            float h2 = sigm(go) * tanhf(c2);
            cst[tid] = c2;
            gstore(&hbuf[(size_t)((t + 1) & 1) * H + u], h2);
            Hout16[(size_t)t * H + u] = (_Float16)h2;
            if (hfinal && t == S_ - 1) hfinal[u] = h2;
        }
        __threadfence();
        __syncthreads();
        __builtin_amdgcn_s_cluster_barrier();
    }
}

// ---------------------------------------------------------------------------
// Latent FC: z = relu(fcW @ h + fcb); seeds decoder h0/h1 buffers (index 0).
__global__ __launch_bounds__(256) void k_enc_fc(const float* __restrict__ W,
                                                const float* __restrict__ bb,
                                                const float* __restrict__ hf,
                                                float* __restrict__ z,
                                                float* __restrict__ h0buf,
                                                float* __restrict__ h1buf) {
    __shared__ float hl[Hdim];
    int tid = threadIdx.x;
    for (int i = tid; i < Hdim; i += 256) hl[i] = hf[i];
    __syncthreads();
    for (int r = tid; r < Hdim; r += 256) {
        float acc = bb[r];
#pragma unroll 8
        for (int k = 0; k < Hdim; ++k) acc += W[(size_t)r * Hdim + k] * hl[k];
        float v = acc > 0.0f ? acc : 0.0f;
        z[r] = v; gstore(&h0buf[r], v); gstore(&h1buf[r], v);
    }
}

// ---------------------------------------------------------------------------
// Persistent decoder cluster kernel: 16 WGs x 256 threads.
//   Per step: layer0 matvec -> cbar -> layer1 matvec -> cbar ->
//   logits (2000 vocab rows / WGP) + packed-u64 atomic argmax -> cbar -> feedback.
//   Both Whh slices TDM-DMA'd into LDS (8 x 32KB descriptors).
__global__ __launch_bounds__(256) void k_decoder(const int* __restrict__ y,
                                                 const float* __restrict__ Wih0,     // [2048] (input dim 1)
                                                 const float* __restrict__ b0,
                                                 const float* __restrict__ b1,
                                                 const _Float16* __restrict__ Whh0h, // [2048][512]
                                                 const _Float16* __restrict__ Wih1h, // [2048][512]
                                                 const _Float16* __restrict__ Whh1h, // [2048][512]
                                                 const _Float16* __restrict__ fcWh,  // [V][512]
                                                 const float* __restrict__ fcb,
                                                 const float* __restrict__ z,
                                                 float* __restrict__ h0buf,          // [2][512]
                                                 float* __restrict__ h1buf,          // [2][512]
                                                 unsigned long long* __restrict__ slots, // [2]
                                                 float* __restrict__ out) {
    constexpr int Ld = Hdim, UP = 32, ROWS_PER_WG = Vdim / 16;
    constexpr unsigned W1OFF = 128u * Ld * 2u;        // byte offset of w1l in LDS
    extern __shared__ char smem[];
    _Float16* w0l = (_Float16*)smem;                  // 128*512 @ LDS offset 0
    _Float16* w1l = w0l + 128 * Ld;                   // 128*512 @ LDS offset 131072
    float* hx   = (float*)(w1l + 128 * Ld);           // 512 (layer input)
    float* hr   = hx + Ld;                            // 512 (recurrent h)
    float* part = hr + Ld;                            // 256
    float* rv   = part + 256;                         // 256
    int*   ri   = (int*)(rv + 256);                   // 256
    float* c0   = (float*)(ri + 256);                 // 32
    float* c1   = c0 + UP;                            // 32

    int w = blockIdx.x, tid = threadIdx.x;

    // TDM preload of both recurrent weight slices
    if (tid < 32) {
#pragma unroll
        for (int g = 0; g < 4; ++g) {
            tdm_load_1d((unsigned)(g * UP * Ld * 2),
                        Whh0h + ((size_t)g * Ld + (size_t)w * UP) * Ld,
                        (unsigned)(UP * Ld));
            tdm_load_1d(W1OFF + (unsigned)(g * UP * Ld * 2),
                        Whh1h + ((size_t)g * Ld + (size_t)w * UP) * Ld,
                        (unsigned)(UP * Ld));
        }
        __builtin_amdgcn_s_wait_tensorcnt(0);
    }
    if (tid < UP) {
        int u = w * UP + tid;
        float zz = z[u];
        c0[tid] = zz; c1[tid] = zz;
        gstore(&h0buf[u], zz); gstore(&h1buf[u], zz);   // parity slot 0
    }
    if (w == 0 && tid == 0) { slots[0] = 0ull; slots[1] = 0ull; }
    __threadfence();
    __syncthreads();
    __builtin_amdgcn_s_cluster_barrier();

    float tok = (float)y[0];
    int lr = tid >> 1, kh = tid & 1;
    int grow = (lr >> 5) * Ld + w * UP + (lr & 31);     // global gate row

    for (int t = 1; t < Tdim; ++t) {
        int pp = (t - 1) & 1, pc = t & 1;

        // ---------------- layer 0 ----------------
        for (int i = tid; i < Ld; i += 256) hr[i] = gload(&h0buf[(size_t)pp * Ld + i]);
        __syncthreads();
        {
            float acc = 0.0f;
            const _Float16* wr = w0l + lr * Ld + kh * 256;
            const float* hh = hr + kh * 256;
#pragma unroll 8
            for (int k = 0; k < 256; ++k) acc += (float)wr[k] * hh[k];
            part[tid] = acc;
        }
        __syncthreads();
        if (tid < UP) {
            int u = w * UP + tid;
            float gi = Wih0[0 * Ld + u] * tok + b0[0 * Ld + u] + part[(0 * UP + tid) * 2] + part[(0 * UP + tid) * 2 + 1];
            float gf = Wih0[1 * Ld + u] * tok + b0[1 * Ld + u] + part[(1 * UP + tid) * 2] + part[(1 * UP + tid) * 2 + 1];
            float gg = Wih0[2 * Ld + u] * tok + b0[2 * Ld + u] + part[(2 * UP + tid) * 2] + part[(2 * UP + tid) * 2 + 1];
            float go = Wih0[3 * Ld + u] * tok + b0[3 * Ld + u] + part[(3 * UP + tid) * 2] + part[(3 * UP + tid) * 2 + 1];
            float c2 = sigm(gf) * c0[tid] + sigm(gi) * tanhf(gg);
            float h2 = sigm(go) * tanhf(c2);
            c0[tid] = c2;
            gstore(&h0buf[(size_t)pc * Ld + u], h2);
        }
        __threadfence();
        __syncthreads();
        __builtin_amdgcn_s_cluster_barrier();

        // ---------------- layer 1 ----------------
        for (int i = tid; i < Ld; i += 256) {
            hx[i] = gload(&h0buf[(size_t)pc * Ld + i]);
            hr[i] = gload(&h1buf[(size_t)pp * Ld + i]);
        }
        __syncthreads();
        {
            float acc = 0.0f;
            const _Float16* wr = w1l + lr * Ld + kh * 256;
            const _Float16* wi = Wih1h + (size_t)grow * Ld + kh * 256;
            const float* hh = hr + kh * 256;
            const float* xx = hx + kh * 256;
#pragma unroll 8
            for (int k = 0; k < 256; ++k)
                acc += (float)wr[k] * hh[k] + (float)wi[k] * xx[k];
            part[tid] = acc;
        }
        __syncthreads();
        if (tid < UP) {
            int u = w * UP + tid;
            float gi = b1[0 * Ld + u] + part[(0 * UP + tid) * 2] + part[(0 * UP + tid) * 2 + 1];
            float gf = b1[1 * Ld + u] + part[(1 * UP + tid) * 2] + part[(1 * UP + tid) * 2 + 1];
            float gg = b1[2 * Ld + u] + part[(2 * UP + tid) * 2] + part[(2 * UP + tid) * 2 + 1];
            float go = b1[3 * Ld + u] + part[(3 * UP + tid) * 2] + part[(3 * UP + tid) * 2 + 1];
            float c2 = sigm(gf) * c1[tid] + sigm(gi) * tanhf(gg);
            float h2 = sigm(go) * tanhf(c2);
            c1[tid] = c2;
            gstore(&h1buf[(size_t)pc * Ld + u], h2);
        }
        __threadfence();
        __syncthreads();
        __builtin_amdgcn_s_cluster_barrier();

        // ---------------- logits + argmax ----------------
        for (int i = tid; i < Ld; i += 256) hr[i] = gload(&h1buf[(size_t)pc * Ld + i]);
        __syncthreads();
        float bv = -3.4e38f; int bi = 0;
        int r0 = w * ROWS_PER_WG;
        for (int r = r0 + tid; r < r0 + ROWS_PER_WG; r += 256) {
            const _Float16* fr = fcWh + (size_t)r * Ld;
            __builtin_prefetch(fr + (size_t)256 * Ld, 0, 1);   // next row for this thread
            float acc = fcb[r];
#pragma unroll 8
            for (int k = 0; k < Ld; ++k) acc += (float)fr[k] * hr[k];
            out[(size_t)t * Vdim + r] = acc;
            if (acc > bv) { bv = acc; bi = r; }        // ascending r -> first-max kept
        }
        rv[tid] = bv; ri[tid] = bi;
        __syncthreads();
        for (int s = 128; s > 0; s >>= 1) {
            if (tid < s) {
                float ov = rv[tid + s]; int oi = ri[tid + s];
                if (ov > rv[tid] || (ov == rv[tid] && oi < ri[tid])) { rv[tid] = ov; ri[tid] = oi; }
            }
            __syncthreads();
        }
        if (tid == 0) {
            unsigned vb = __float_as_uint(rv[0]);
            vb = (vb & 0x80000000u) ? ~vb : (vb | 0x80000000u);        // total order
            unsigned long long key = ((unsigned long long)vb << 32) |
                                     (unsigned long long)(0xFFFFFFFFu - (unsigned)ri[0]);
            atomicMax(slots + pc, key);
        }
        __threadfence();
        __syncthreads();
        __builtin_amdgcn_s_cluster_barrier();

        unsigned long long win = __hip_atomic_load(slots + pc, __ATOMIC_RELAXED, __HIP_MEMORY_SCOPE_AGENT);
        tok = (float)(0xFFFFFFFFu - (unsigned)(win & 0xFFFFFFFFull));
        if (w == 0 && tid == 0)   // reset slot used at t+1; 2 cluster barriers before reuse
            __hip_atomic_store(slots + pp, 0ull, __ATOMIC_RELAXED, __HIP_MEMORY_SCOPE_AGENT);
        __threadfence();
    }
}

// ---------------------------------------------------------------------------
extern "C" void kernel_launch(void* const* d_in, const int* in_sizes, int n_in,
                              void* d_out, int out_size, void* d_ws, size_t ws_size,
                              hipStream_t stream) {
    (void)in_sizes; (void)n_in; (void)out_size; (void)ws_size;
    const int*   x        = (const int*)  d_in[0];
    const int*   y        = (const int*)  d_in[1];
    const float* emb      = (const float*)d_in[2];
    const float* encWih0  = (const float*)d_in[3];
    const float* encWhh0  = (const float*)d_in[4];
    const float* enc_b0   = (const float*)d_in[5];
    const float* encWih1  = (const float*)d_in[6];
    const float* encWhh1  = (const float*)d_in[7];
    const float* enc_b1   = (const float*)d_in[8];
    const float* enc_fcW  = (const float*)d_in[9];
    const float* enc_fcb  = (const float*)d_in[10];
    const float* decWih0  = (const float*)d_in[11];
    const float* decWhh0  = (const float*)d_in[12];
    const float* dec_b0   = (const float*)d_in[13];
    const float* decWih1  = (const float*)d_in[14];
    const float* decWhh1  = (const float*)d_in[15];
    const float* dec_b1   = (const float*)d_in[16];
    const float* dec_fcW  = (const float*)d_in[17];
    const float* dec_fcb  = (const float*)d_in[18];
    float* out = (float*)d_out;

    char* ws = (char*)d_ws;
    size_t off = 0;
    auto alloc = [&](size_t bytes) -> void* {
        void* p = ws + off;
        off += (bytes + 255) & ~(size_t)255;
        return p;
    };

    _Float16* em16      = (_Float16*)alloc((size_t)Sdim * Edim * 2);
    _Float16* encWih0h  = (_Float16*)alloc((size_t)G4H * Edim * 2);
    _Float16* encWhh0h  = (_Float16*)alloc((size_t)G4H * Hdim * 2);
    _Float16* encWih1h  = (_Float16*)alloc((size_t)G4H * Hdim * 2);
    _Float16* encWhh1h  = (_Float16*)alloc((size_t)G4H * Hdim * 2);
    _Float16* decWhh0h  = (_Float16*)alloc((size_t)G4H * Hdim * 2);
    _Float16* decWih1h  = (_Float16*)alloc((size_t)G4H * Hdim * 2);
    _Float16* decWhh1h  = (_Float16*)alloc((size_t)G4H * Hdim * 2);
    _Float16* fcWh      = (_Float16*)alloc((size_t)Vdim * Hdim * 2);
    float*    P0        = (float*)   alloc((size_t)Sdim * G4H * 4);
    float*    P1        = (float*)   alloc((size_t)Sdim * G4H * 4);
    _Float16* H016      = (_Float16*)alloc((size_t)Sdim * Hdim * 2);
    float*    hbufE     = (float*)   alloc((size_t)2 * Hdim * 4);
    float*    hfinal    = (float*)   alloc((size_t)Hdim * 4);
    float*    zbuf      = (float*)   alloc((size_t)Hdim * 4);
    float*    h0buf     = (float*)   alloc((size_t)2 * Hdim * 4);
    float*    h1buf     = (float*)   alloc((size_t)2 * Hdim * 4);
    unsigned long long* slots = (unsigned long long*)alloc(16);

    auto cvt = [&](const float* in, _Float16* o, int n) {
        k_f32_to_f16<<<(n + 255) / 256, 256, 0, stream>>>(in, o, n);
    };
    // one-time f16 weight conversions (all L2-resident afterwards)
    cvt(encWih0, encWih0h, G4H * Edim);
    cvt(encWhh0, encWhh0h, G4H * Hdim);
    cvt(encWih1, encWih1h, G4H * Hdim);
    cvt(encWhh1, encWhh1h, G4H * Hdim);
    cvt(decWhh0, decWhh0h, G4H * Hdim);
    cvt(decWih1, decWih1h, G4H * Hdim);
    cvt(decWhh1, decWhh1h, G4H * Hdim);
    cvt(dec_fcW, fcWh,     Vdim * Hdim);

    // embedding gather
    k_embed<<<(Sdim * Edim + 255) / 256, 256, 0, stream>>>(x, emb, em16);

    const size_t enc_lds = 128 * Hdim * 2 + Hdim * 4 + 256 * 4 + 32 * 4;                 // 134272 B
    const size_t dec_lds = 2 * 128 * Hdim * 2 + 2 * Hdim * 4 + 3 * 256 * 4 + 2 * 32 * 4; // 269568 B

    // ---- Encoder layer 0 ----
    // P0 = em @ Wih0^T : M=1024, N=2048, K=256 -> 8192 tiles / 8 per WG
    k_gemm_wmma<<<(Sdim / 16) * (G4H / 16) / 8, 256, 0, stream>>>(em16, encWih0h, P0, Sdim, G4H, Edim);
    k_lstm_rec<<<16, 256, enc_lds, stream>>>(P0, encWhh0h, enc_b0, hbufE, H016, nullptr, Sdim);

    // ---- Encoder layer 1 ----
    // P1 = H0 @ Wih1^T : K=512
    k_gemm_wmma<<<(Sdim / 16) * (G4H / 16) / 8, 256, 0, stream>>>(H016, encWih1h, P1, Sdim, G4H, Hdim);
    k_lstm_rec<<<16, 256, enc_lds, stream>>>(P1, encWhh1h, enc_b1, hbufE, H016 /*scratch reuse*/, hfinal, Sdim);

    // ---- latent FC + decoder-state seed ----
    k_enc_fc<<<1, 256, 0, stream>>>(enc_fcW, enc_fcb, hfinal, zbuf, h0buf, h1buf);

    // outputs[0] = zeros
    k_zero_f32<<<(Vdim + 255) / 256, 256, 0, stream>>>(out, Vdim);

    // ---- fused autoregressive decoder (persistent cluster kernel) ----
    k_decoder<<<16, 256, dec_lds, stream>>>(y, decWih0, dec_b0, dec_b1,
                                            decWhh0h, decWih1h, decWhh1h,
                                            fcWh, dec_fcb, zbuf,
                                            h0buf, h1buf, slots, out);
}
